// HashingMemory_53163105190602
// MI455X (gfx1250) — matrix operand order repaired and checked by reading, not compiled, not run on hardware
//
#include <hip/hip_runtime.h>

#define HEADS   4
#define KNN     32
#define N_KEYS  512
#define K_DIM   512
#define HALF    256
#define V_DIM   1024
#define IN_DIM  1024
#define SEQ     2048
#define NQ      2048   // HEADS*K_DIM

typedef __attribute__((ext_vector_type(16))) __bf16 v16bf;
typedef __attribute__((ext_vector_type(8)))  __bf16 v8bf;
typedef __attribute__((ext_vector_type(8)))  float  v8f;
typedef __attribute__((ext_vector_type(4)))  float  v4f;

__device__ __forceinline__ v16bf cat16(v8bf lo, v8bf hi) {
    return __builtin_shufflevector(lo, hi, 0, 1, 2, 3, 4, 5, 6, 7,
                                           8, 9, 10, 11, 12, 13, 14, 15);
}

#define WMMA4(A, B)                                                            \
    _Pragma("unroll")                                                          \
    for (int j = 0; j < 4; ++j)                                                \
        acc[j] = __builtin_amdgcn_wmma_f32_16x16x32_bf16(                      \
                     false, (A), false, (B)[j], (short)0, acc[j], false, false);

// ---------------------------------------------------------------------------
// Kernel 0: fp32 -> bf16 pack (streamed, 16B stores)
// ---------------------------------------------------------------------------
__global__ __launch_bounds__(256)
void cvt_bf16_kernel(const float* __restrict__ src, __bf16* __restrict__ dst)
{
    int i = (blockIdx.x * 256 + threadIdx.x) * 8;
    v8bf o;
#pragma unroll
    for (int e = 0; e < 8; ++e) o[e] = (__bf16)src[i + e];
    *(v8bf*)(dst + i) = o;
}

// ---------------------------------------------------------------------------
// Kernel 1: q = x @ w_q^T + b_q  (2048x2048x1024). Each wave: 16x64 strip.
// 2x-unrolled ping-pong double-buffer: no swap moves, loads overlap WMMAs.
// ---------------------------------------------------------------------------
__global__ __launch_bounds__(256)
void qproj_kernel(const __bf16* __restrict__ xb, const __bf16* __restrict__ wqb,
                  const float* __restrict__ bq, __bf16* __restrict__ q)
{
    int wave = threadIdx.x >> 5;
    int lane = threadIdx.x & 31;
    int lrow = lane & 15;
    int g    = lane >> 4;

    int id = blockIdx.x * 8 + wave;        // 0..4095
    int tm = id >> 5;                      // 128 M-tiles
    int ng = id & 31;                      // 32 N-groups of 64

    int arow = tm * 16 + lrow;
    const __bf16* xrow = xb + (size_t)arow * IN_DIM;

#define Q_LOAD_FRAGS(KB, A, B)                                                 \
    {                                                                          \
        const v8bf* xr = (const v8bf*)(xrow + (KB));                           \
        A = cat16(xr[g], xr[2 + g]);            /* K: g*8..+7, 16+g*8..+7 */   \
        _Pragma("unroll")                                                      \
        for (int j = 0; j < 4; ++j) {                                          \
            const v8bf* wr = (const v8bf*)(wqb +                               \
                (size_t)(ng * 64 + j * 16 + lrow) * IN_DIM + (KB));            \
            B[j] = cat16(wr[g], wr[2 + g]);     /* K: g*8..+7, 16+g*8..+7 */   \
        }                                                                      \
    }

    v8f acc[4] = {};
    v16bf aP, bP[4], aQ, bQ[4];
    Q_LOAD_FRAGS(0, aP, bP);

    const int KSTEPS = IN_DIM / 32;        // 32 (even)
    for (int ks = 0; ks < KSTEPS - 2; ks += 2) {
        Q_LOAD_FRAGS((ks + 1) * 32, aQ, bQ);   // in flight under WMMA(P)
        WMMA4(aP, bP);
        Q_LOAD_FRAGS((ks + 2) * 32, aP, bP);   // in flight under WMMA(Q)
        WMMA4(aQ, bQ);
    }
    Q_LOAD_FRAGS((KSTEPS - 1) * 32, aQ, bQ);
    WMMA4(aP, bP);
    WMMA4(aQ, bQ);
#undef Q_LOAD_FRAGS

#pragma unroll
    for (int j = 0; j < 4; ++j) {
        int n = ng * 64 + j * 16 + lrow;
        float bias = bq[n];
#pragma unroll
        for (int r = 0; r < 8; ++r) {
            int m = tm * 16 + g * 8 + r;             // C layout: vgpr r, half g
            q[(size_t)m * NQ + n] = (__bf16)(acc[j][r] + bias);
        }
    }
}

// ---------------------------------------------------------------------------
// Kernel 2: scores (16 tokens x 512 keys, K=256). q-tile staged in LDS via
// async global->LDS b128 (ASYNCcnt); global B fragments ping-pong buffered;
// WMMA into LDS scores, then per-token top-32.
// ---------------------------------------------------------------------------
__global__ __launch_bounds__(256)
void scores_topk_kernel(const __bf16* __restrict__ q, const __bf16* __restrict__ keysb,
                        float* __restrict__ s_top, int* __restrict__ i_top)
{
    __shared__ __align__(16) __bf16 qs[16][HALF];   // 8 KB (shared A tile)
    __shared__ float sc[16][512];                   // 32 KB (scores)

    int tid  = threadIdx.x;
    int wave = tid >> 5;
    int lane = tid & 31;
    int lrow = lane & 15;
    int g    = lane >> 4;

    int ht   = blockIdx.y;                 // h*2 + t
    int h    = ht >> 1, t = ht & 1;
    int tok0 = blockIdx.x * 16;

    const __bf16* qbase = q + (size_t)tok0 * NQ + h * K_DIM + t * HALF;

    // Stage 16x256 bf16 q-tile: 512 x 16B chunks, 2 per thread, async to LDS.
#pragma unroll
    for (int i = 0; i < 2; ++i) {
        int c   = tid + i * 256;
        int row = c >> 5;
        int c8  = c & 31;
        unsigned lds = (unsigned)(uintptr_t)&qs[row][c8 * 8];
        unsigned long long ga =
            (unsigned long long)(uintptr_t)(qbase + (size_t)row * NQ + c8 * 8);
        asm volatile("global_load_async_to_lds_b128 %0, %1, off"
                     :: "v"(lds), "v"(ga) : "memory");
    }
    asm volatile("s_wait_asynccnt 0" ::: "memory");
    __syncthreads();

    const __bf16* kbase = keysb + (size_t)ht * N_KEYS * HALF;
    int nt0 = wave * 4;                    // 4 consecutive key-tiles per wave

#define S_LOAD_B(KB, B)                                                        \
    {                                                                          \
        _Pragma("unroll")                                                      \
        for (int j = 0; j < 4; ++j) {                                          \
            const v8bf* kr = (const v8bf*)(kbase +                             \
                (size_t)((nt0 + j) * 16 + lrow) * HALF + (KB));                \
            B[j] = cat16(kr[g], kr[2 + g]);                                    \
        }                                                                      \
    }
#define S_LOAD_A(KS, A)                                                        \
    {                                                                          \
        const v8bf* qr = (const v8bf*)(&qs[lrow][(KS) * 32]);                  \
        A = cat16(qr[g], qr[2 + g]);            /* ds_load_b128 x2 */          \
    }

    v8f acc[4] = {};
    v16bf a, bP[4], bQ[4];
    S_LOAD_B(0, bP);

    const int KSTEPS = HALF / 32;          // 8 (even)
    for (int ks = 0; ks < KSTEPS - 2; ks += 2) {
        S_LOAD_B((ks + 1) * 32, bQ);       // global loads overlap WMMA(P)
        S_LOAD_A(ks, a);
        WMMA4(a, bP);
        S_LOAD_B((ks + 2) * 32, bP);       // global loads overlap WMMA(Q)
        S_LOAD_A(ks + 1, a);
        WMMA4(a, bQ);
    }
    S_LOAD_B((KSTEPS - 1) * 32, bQ);
    S_LOAD_A(KSTEPS - 2, a);
    WMMA4(a, bP);
    S_LOAD_A(KSTEPS - 1, a);
    WMMA4(a, bQ);
#undef S_LOAD_B
#undef S_LOAD_A

#pragma unroll
    for (int j = 0; j < 4; ++j)
#pragma unroll
        for (int r = 0; r < 8; ++r)
            sc[g * 8 + r][(nt0 + j) * 16 + lrow] = acc[j][r];
    __syncthreads();

    // top-32 per token; 2 tokens per wave; lane scans 16 columns
    for (int rep = 0; rep < 2; ++rep) {
        int tloc = wave * 2 + rep;
        volatile float* row = (volatile float*)&sc[tloc][0];
        int c0 = lane * 16;
        for (int it = 0; it < KNN; ++it) {
            float best = -3.4e38f; int bidx = 0x7fffffff;
#pragma unroll 4
            for (int c = 0; c < 16; ++c) {
                float v = row[c0 + c];
                if (v > best || (v == best && (c0 + c) < bidx)) { best = v; bidx = c0 + c; }
            }
            for (int off = 16; off; off >>= 1) {           // wave32 butterfly
                float ov = __shfl_xor(best, off);
                int   oi = __shfl_xor(bidx, off);
                if (ov > best || (ov == best && oi < bidx)) { best = ov; bidx = oi; }
            }
            if (lane == 0) {
                size_t o = ((((size_t)(tok0 + tloc) * HEADS + h) * 2 + t) * KNN + it);
                s_top[o] = best;
                i_top[o] = bidx;
                row[bidx] = -3.4e38f;
            }
        }
    }
}

// ---------------------------------------------------------------------------
// Kernel 3: combine halves (32x32 sums), top-32 of 1024, softmax.
// one wave32 per (token, head) -- 32 lanes == 32 slots
// ---------------------------------------------------------------------------
__global__ __launch_bounds__(256)
void combine_kernel(const float* __restrict__ s_top, const int* __restrict__ i_top,
                    float* __restrict__ wsel, int* __restrict__ isel)
{
    __shared__ float s2sh[8][32];
    __shared__ int   i2sh[8][32];

    int wave = threadIdx.x >> 5;
    int lane = threadIdx.x & 31;
    int id   = blockIdx.x * 8 + wave;      // token*HEADS + h
    int b    = id >> 2, h = id & 3;

    size_t base = (((size_t)b * HEADS + h) * 2) * KNN;
    float s1v = s_top[base + lane];
    int   i1v = i_top[base + lane];
    s2sh[wave][lane] = s_top[base + KNN + lane];
    i2sh[wave][lane] = i_top[base + KNN + lane];
    __syncthreads();

    unsigned taken = 0u;
    float sel_s = 0.f; int sel_i = 0; float smax = 0.f;

    for (int it = 0; it < KNN; ++it) {
        float best = -3.4e38f; int bcol = 0;
        for (int j = 0; j < 32; ++j) {
            if (!(taken & (1u << j))) {
                float v = s1v + s2sh[wave][j];
                if (v > best) { best = v; bcol = j; }
            }
        }
        int brow = lane;
        for (int off = 16; off; off >>= 1) {
            float ov  = __shfl_xor(best, off);
            int   orw = __shfl_xor(brow, off);
            int   ocl = __shfl_xor(bcol, off);
            if (ov > best ||
                (ov == best && (orw * 32 + ocl) < (brow * 32 + bcol))) {
                best = ov; brow = orw; bcol = ocl;
            }
        }
        if (lane == brow) taken |= (1u << bcol);
        int i1r  = __shfl(i1v, brow);
        int pair = i1r * N_KEYS + i2sh[wave][bcol];
        if (it == 0) smax = best;
        if (lane == it) { sel_s = best; sel_i = pair; }
    }

    float e = __expf(sel_s - smax);
    float sum = e;
    for (int off = 16; off; off >>= 1) sum += __shfl_xor(sum, off);
    float wv = e / sum;

    size_t o = (size_t)b * (HEADS * KNN) + h * KNN + lane;
    wsel[o] = wv;
    isel[o] = sel_i;
}

// ---------------------------------------------------------------------------
// Kernel 4: out[b] = sum_k w[k] * values[idx[k]]  -- HBM-bound gather.
// Non-temporal 128-bit loads (1 GB table, ~zero reuse -> keep L2 clean).
// ---------------------------------------------------------------------------
__global__ __launch_bounds__(256)
void gather_kernel(const float* __restrict__ values, const float* __restrict__ wsel,
                   const int* __restrict__ isel, float* __restrict__ out)
{
    __shared__ float wsh[HEADS * KNN];
    __shared__ int   ish[HEADS * KNN];

    int tok = blockIdx.x;
    if (threadIdx.x < HEADS * KNN) {
        wsh[threadIdx.x] = wsel[(size_t)tok * (HEADS * KNN) + threadIdx.x];
        ish[threadIdx.x] = isel[(size_t)tok * (HEADS * KNN) + threadIdx.x];
    }
    __syncthreads();

    int d0 = threadIdx.x * 4;
    float a0 = 0.f, a1 = 0.f, a2 = 0.f, a3 = 0.f;
    for (int k = 0; k < HEADS * KNN; ++k) {
        if (k + 1 < HEADS * KNN)
            __builtin_prefetch(values + (size_t)ish[k + 1] * V_DIM + d0, 0, 1);
        v4f v = __builtin_nontemporal_load(
                    (const v4f*)(values + (size_t)ish[k] * V_DIM + d0));
        float wk = wsh[k];
        a0 += wk * v.x; a1 += wk * v.y; a2 += wk * v.z; a3 += wk * v.w;
    }
    v4f o; o.x = a0; o.y = a1; o.z = a2; o.w = a3;
    __builtin_nontemporal_store(o, (v4f*)(out + (size_t)tok * V_DIM + d0));
}

// ---------------------------------------------------------------------------
extern "C" void kernel_launch(void* const* d_in, const int* in_sizes, int n_in,
                              void* d_out, int out_size, void* d_ws, size_t ws_size,
                              hipStream_t stream) {
    (void)in_sizes; (void)n_in; (void)out_size; (void)ws_size;

    const float* x      = (const float*)d_in[0];
    const float* wq     = (const float*)d_in[1];
    const float* bq     = (const float*)d_in[2];
    const float* keys   = (const float*)d_in[3];
    const float* values = (const float*)d_in[4];
    float*       out    = (float*)d_out;

    char* ws = (char*)d_ws;
    __bf16* q     = (__bf16*)(ws);                 // 2048*2048*2 = 8 MB
    __bf16* xb    = (__bf16*)(ws + (8u  << 20));   // 4 MB
    __bf16* wqb   = (__bf16*)(ws + (12u << 20));   // 4 MB
    __bf16* keysb = (__bf16*)(ws + (16u << 20));   // 2 MB
    float*  s_top = (float*) (ws + (18u << 20));   // 2 MB
    int*    i_top = (int*)   (ws + (20u << 20));   // 2 MB
    float*  wsel  = (float*) (ws + (22u << 20));   // 1 MB
    int*    isel  = (int*)   (ws + (23u << 20));   // 1 MB

    // 0) bf16 packs
    cvt_bf16_kernel<<<(SEQ * IN_DIM) / (8 * 256), 256, 0, stream>>>(x, xb);
    cvt_bf16_kernel<<<(NQ * IN_DIM) / (8 * 256), 256, 0, stream>>>(wq, wqb);
    cvt_bf16_kernel<<<(HEADS * 2 * N_KEYS * HALF) / (8 * 256), 256, 0, stream>>>(keys, keysb);

    // 1) q projection: 4096 waves (16x64 strips), 8 waves/block
    qproj_kernel<<<512, 256, 0, stream>>>(xb, wqb, bq, q);

    // 2) scores + per-half top-32
    scores_topk_kernel<<<dim3(128, 8), 256, 0, stream>>>(q, keysb, s_top, i_top);

    // 3) combine + softmax
    combine_kernel<<<1024, 256, 0, stream>>>(s_top, i_top, wsel, isel);

    // 4) value gather (HBM-bound)
    gather_kernel<<<2048, 256, 0, stream>>>(values, wsel, isel, out);
}
